// SequenceBRNN_58102317580334
// MI455X (gfx1250) — matrix operand order repaired and checked by reading, not compiled, not run on hardware
//
#include <hip/hip_runtime.h>
#include <hip/hip_bf16.h>

typedef __bf16 bf16_t;
typedef bf16_t v16bf __attribute__((ext_vector_type(16)));
typedef bf16_t v8bf  __attribute__((ext_vector_type(8)));
typedef bf16_t v4bf  __attribute__((ext_vector_type(4)));
typedef float  v8f   __attribute__((ext_vector_type(8)));
typedef float  v4f   __attribute__((ext_vector_type(4)));

#define B_  512
#define T_  512
#define F_  128
#define U_  256
#define N4_ 1024

// Packed bf16 weight layout in workspace (elements): per direction
//   [Wt0 1024x128][Ut0 1024x256][Wt1 1024x256][Ut1][Wt2][Ut2], all [N][K]
#define DIRSTRIDE 1441792ull

struct BiasPtrs { const float* b[2][3]; };

__device__ __forceinline__ float sigf(float x)   { return 1.0f / (1.0f + __expf(-x)); }
__device__ __forceinline__ float tanhfst(float x){ return 2.0f / (1.0f + __expf(-2.0f * x)) - 1.0f; }

// f32 [K][N] row-major  ->  bf16 [N][K] (transposed) for WMMA B-fragments
__global__ __launch_bounds__(256) void cvt_transpose_kernel(
    const float* __restrict__ src, bf16_t* __restrict__ dst, int K, int N)
{
    int idx = blockIdx.x * 256 + threadIdx.x;
    if (idx >= K * N) return;
    int n = idx / K;
    int k = idx - n * K;
    dst[idx] = (bf16_t)src[k * N + n];
}

// acc += A(16xK, LDS) x W(KxN tiles, bf16 [N][K] streamed from L2).
// Software-pipelined: slab s+1's 8 B-fragments are issued before slab s's
// 8 WMMAs, so every WMMA waits on loads issued a full slab earlier.
template <int K>
__device__ __forceinline__ void gemm_accum(
    const bf16_t* __restrict__ aLds, int ldA,
    const bf16_t* __restrict__ wt, int lane, int wave, v8f acc[4][2])
{
    const int l15  = lane & 15;
    const int aoff = (lane & 16) ? 16 : 0;   // A frag: lanes>=16 hold K+16..31
    const int boff = (lane & 16) ? 16 : 0;   // B frag: lanes>=16 hold K+16..31
    const bf16_t* arow = aLds + l15 * ldA;
    // Per-lane column base; per-fragment deltas are compile-time constants
    // that fold into the instruction's immediate offset.
    const bf16_t* wcol = wt + (size_t)(wave * 32 + l15) * K + boff;

    constexpr int NS = K / 32;
    v16bf buf0[8], buf1[8];

#pragma unroll
    for (int i = 0; i < 8; ++i) {
        int g = i >> 1, h = i & 1;
        buf0[i] = *(const v16bf*)(wcol + (size_t)(g * U_ + h * 16) * K);
    }

#pragma unroll
    for (int s = 0; s < NS; ++s) {
        const int kb = s * 32;
        v16bf* cur = (s & 1) ? buf1 : buf0;
        v16bf* nxt = (s & 1) ? buf0 : buf1;
        if (s + 1 < NS) {
#pragma unroll
            for (int i = 0; i < 8; ++i) {
                int g = i >> 1, h = i & 1;
                nxt[i] = *(const v16bf*)(wcol + (size_t)(g * U_ + h * 16) * K + kb + 32);
            }
        }
        v8bf alo = *(const v8bf*)(arow + kb + aoff);
        v8bf ahi = *(const v8bf*)(arow + kb + aoff + 8);
        v16bf afrag = __builtin_shufflevector(alo, ahi,
            0, 1, 2, 3, 4, 5, 6, 7, 8, 9, 10, 11, 12, 13, 14, 15);
#pragma unroll
        for (int i = 0; i < 8; ++i) {
            int g = i >> 1, h = i & 1;
            acc[g][h] = __builtin_amdgcn_wmma_f32_16x16x32_bf16(
                false, afrag, false, cur[i], (short)0, acc[g][h], false, false);
        }
    }
}

// Persistent kernel: one workgroup = (direction, 16-row batch tile), runs all
// T=512 steps of the 3-cell stack.  Weights stream from L2 every step.
// To keep VGPR pressure low (so the B-fragment pipeline survives register
// allocation), c-state and biases live in LDS; each lane touches only its own
// c elements, so no extra barriers are needed.
__global__ __launch_bounds__(256) void lstm_persistent_kernel(
    const float* __restrict__ x,      // [B][T][F] f32
    const bf16_t* __restrict__ wts,   // packed bf16 weights
    BiasPtrs bp,
    float* __restrict__ hfin)         // [2][B][U] f32 (final h of top cell)
{
    __shared__ __align__(64) bf16_t sX[16][F_];        // staged x_t tile (bf16)    4KB
    __shared__ __align__(64) bf16_t sH[3][16][U_];     // per-cell hidden state    24KB
    __shared__ __align__(64) float  sC[3][16][U_];     // per-cell c-state (f32)   48KB
    __shared__ __align__(64) float  sBias[3][N4_];     // per-cell gate biases     12KB

    const int dir  = blockIdx.x >> 5;          // 0 = fwd, 1 = bwd
    const int tile = blockIdx.x & 31;          // 16-row batch tile
    const int tid  = threadIdx.x;
    const int wave = tid >> 5;
    const int lane = tid & 31;
    const int l15  = lane & 15;
    const int rhi  = (lane & 16) ? 8 : 0;      // C/D tile: lanes>=16 hold M=8..15

    const unsigned long long wOff[3] = {0ull, 393216ull, 917504ull};
    const unsigned long long uOff[3] = {131072ull, 655360ull, 1179648ull};

    // Zero h and c; stage biases into LDS.
    for (int i = tid; i < 3 * 16 * U_; i += 256) {
        ((bf16_t*)sH)[i] = (bf16_t)0.0f;
        ((float*)sC)[i]  = 0.0f;
    }
    for (int i = tid; i < 3 * N4_; i += 256) {
        int c = i >> 10, n = i & (N4_ - 1);
        sBias[c][n] = bp.b[dir][c][n];
    }
    __syncthreads();

    const float* xbase = x + (size_t)(tile * 16) * T_ * F_;

#pragma clang loop unroll(disable)
    for (int t = 0; t < T_; ++t) {
        // Opaque scalar offset, redefined every step: blocks LICM from hoisting
        // 350+ weight fragments out of the t-loop (which previously spilled to
        // scratch), WITHOUT laundering the pointer through an integer (which
        // previously degraded global_load to flat_load).
        unsigned int zoff = 0;
        asm volatile("" : "+s"(zoff));
        const bf16_t* wdir = wts + (size_t)dir * DIRSTRIDE + zoff;

        const int tt = dir ? (T_ - 1 - t) : t;   // bwd direction scans reversed x
        // Stage x_t tile (f32 -> bf16); non-temporal: x is single-use, keep it
        // from evicting the hot 5.8MB weight set out of L2.  Also prefetch the
        // next step's tile (global_prefetch_b8), one probe per 64B line.
        const float* xt = xbase + (size_t)tt * F_;
        const int tt2 = dir ? (tt - 1) : (tt + 1);
        const float* xt2 = xbase + (size_t)(tt2 < 0 ? 0 : (tt2 >= T_ ? T_ - 1 : tt2)) * F_;
        for (int j = tid; j < 16 * 32; j += 256) {      // 512 float4 chunks
            int r = j >> 5, k4 = (j & 31) * 4;
            v4f v = __builtin_nontemporal_load((const v4f*)(xt + (size_t)r * T_ * F_ + k4));
            if ((j & 3) == 0)
                __builtin_prefetch(xt2 + (size_t)r * T_ * F_ + k4, 0, 1);
            v4bf o;
            o[0] = (bf16_t)v[0]; o[1] = (bf16_t)v[1];
            o[2] = (bf16_t)v[2]; o[3] = (bf16_t)v[3];
            *(v4bf*)&sX[r][k4] = o;
        }
        __syncthreads();

#pragma unroll
        for (int cell = 0; cell < 3; ++cell) {
            // Init accumulator with the bias (uniform across the 8 rows).
            v8f acc[4][2];
#pragma unroll
            for (int g = 0; g < 4; ++g)
#pragma unroll
                for (int h = 0; h < 2; ++h) {
                    float bv = sBias[cell][g * U_ + wave * 32 + h * 16 + l15];
#pragma unroll
                    for (int r = 0; r < 8; ++r) acc[g][h][r] = bv;
                }

            if (cell == 0)
                gemm_accum<F_>(&sX[0][0], F_, wdir + wOff[0], lane, wave, acc);          // x @ W0
            else
                gemm_accum<U_>(&sH[cell - 1][0][0], U_, wdir + wOff[cell], lane, wave, acc); // h_low @ W
            gemm_accum<U_>(&sH[cell][0][0], U_, wdir + uOff[cell], lane, wave, acc);     // h @ U

            __syncthreads();   // everyone done reading old sH[cell]

            // Gate fusion: all four gates for a unit live in the same lane/slot
#pragma unroll
            for (int h = 0; h < 2; ++h) {
                const int unit = wave * 32 + h * 16 + l15;
#pragma unroll
                for (int r = 0; r < 8; ++r) {
                    const int row = r + rhi;
                    float cprev = sC[cell][row][unit];
                    float cn = sigf(acc[1][h][r]) * cprev
                             + sigf(acc[0][h][r]) * tanhfst(acc[2][h][r]);
                    sC[cell][row][unit] = cn;
                    float hv = sigf(acc[3][h][r]) * tanhfst(cn);
                    sH[cell][row][unit] = (bf16_t)hv;
                    if (cell == 2 && t == T_ - 1) {
                        hfin[((size_t)dir * B_ + tile * 16 + row) * U_ + unit] = hv;
                    }
                }
            }
            __syncthreads();   // new sH[cell] visible before next consumer
        }
    }
}

// FC head: out[b] = sigmoid( relu(concat(h_f,h_b) @ W_fc1) @ W_fc2 )
__global__ __launch_bounds__(256) void fc_kernel(
    const float* __restrict__ hfin,   // [2][B][U]
    const float* __restrict__ Wfc1,   // [512][512]
    const float* __restrict__ Wfc2,   // [512][1]
    float* __restrict__ out)          // [B]
{
    __shared__ float sh[512];
    __shared__ float red[256];
    const int b = blockIdx.x, tid = threadIdx.x;

    for (int i = tid; i < 512; i += 256) {
        int d = i >> 8, u = i & 255;
        sh[i] = hfin[((size_t)d * B_ + b) * U_ + u];
    }
    __syncthreads();

    float a0 = 0.0f, a1 = 0.0f;
    const int j0 = tid, j1 = tid + 256;
    for (int k = 0; k < 512; ++k) {
        float hk = sh[k];
        const float* wr = Wfc1 + (size_t)k * 512;
        a0 = fmaf(hk, wr[j0], a0);
        a1 = fmaf(hk, wr[j1], a1);
    }
    red[tid] = fmaxf(a0, 0.0f) * Wfc2[j0] + fmaxf(a1, 0.0f) * Wfc2[j1];
    __syncthreads();
    for (int s = 128; s > 0; s >>= 1) {
        if (tid < s) red[tid] += red[tid + s];
        __syncthreads();
    }
    if (tid == 0) out[b] = 1.0f / (1.0f + __expf(-red[0]));
}

extern "C" void kernel_launch(void* const* d_in, const int* in_sizes, int n_in,
                              void* d_out, int out_size, void* d_ws, size_t ws_size,
                              hipStream_t stream)
{
    (void)in_sizes; (void)n_in; (void)out_size; (void)ws_size;
    const float* x = (const float*)d_in[0];

    bf16_t* wts = (bf16_t*)d_ws;
    const size_t wOff[3] = {0ull, 393216ull, 917504ull};
    const size_t uOff[3] = {131072ull, 655360ull, 1179648ull};

    BiasPtrs bp;
    for (int d = 0; d < 2; ++d) {
        for (int c = 0; c < 3; ++c) {
            const int base = 1 + d * 9 + c * 3;           // W, U, b per cell
            const float* W  = (const float*)d_in[base];
            const float* Uw = (const float*)d_in[base + 1];
            bp.b[d][c]      = (const float*)d_in[base + 2];
            const int Kin = (c == 0) ? F_ : U_;
            int totW = Kin * N4_;
            cvt_transpose_kernel<<<(totW + 255) / 256, 256, 0, stream>>>(
                W, wts + d * DIRSTRIDE + wOff[c], Kin, N4_);
            int totU = U_ * N4_;
            cvt_transpose_kernel<<<(totU + 255) / 256, 256, 0, stream>>>(
                Uw, wts + d * DIRSTRIDE + uOff[c], U_, N4_);
        }
    }

    float* hfin = (float*)((char*)d_ws + 2ull * DIRSTRIDE * sizeof(bf16_t));

    lstm_persistent_kernel<<<64, 256, 0, stream>>>(x, wts, bp, hfin);

    fc_kernel<<<B_, 256, 0, stream>>>(
        hfin, (const float*)d_in[19], (const float*)d_in[20], (float*)d_out);
}
